// GNSModel_47425028883052
// MI455X (gfx1250) — matrix operand order repaired and checked by reading, not compiled, not run on hardware
//
#include <hip/hip_runtime.h>

// ---------------------------------------------------------------------------
// GNS forward on gfx1250 (MI455X). Compute-bound (~220 GFLOP of 128-wide GEMM)
// -> all GEMMs on v_wmma_f32_16x16x32_bf16 (f32 accum).
//  * Weights pre-packed to bf16 B-fragment layout, staged into LDS once per
//    256-thread block (8 waves share) -- via TDM TENSOR_LOAD_TO_LDS when the
//    builtin exists (TENSORcnt path), else cooperative b128 copy.
//  * bf16 mirrors of h/e in global memory: A-fragments load as packed dwords.
//    fp32 masters kept for exact residuals and the scatter-sum.
//  * Native (__bf16) casts so the backend emits hw cvt ops, not integer RNE.
// ---------------------------------------------------------------------------

#define NND   50000
#define EED   400000
#define HD    128
#define SSTEP 3
#define NODE_IN 4
#define EDGE_IN 3
#define OUT_D 2
#define ACTW  136   // padded bf16 LDS activation row stride (68 dwords)
#define ACTDW 68

typedef __bf16 bf16_t;
typedef bf16_t v16bf __attribute__((ext_vector_type(16)));
typedef bf16_t v2bf  __attribute__((ext_vector_type(2)));
typedef float  v8f   __attribute__((ext_vector_type(8)));
typedef unsigned int v8u __attribute__((ext_vector_type(8)));
typedef unsigned int u32x4 __attribute__((ext_vector_type(4)));
typedef int i32x8 __attribute__((ext_vector_type(8)));
typedef int i32x4 __attribute__((ext_vector_type(4)));

#if __has_builtin(__builtin_amdgcn_tensor_load_to_lds) && \
    __has_builtin(__builtin_amdgcn_s_wait_tensorcnt)
#define HAVE_TDM 1
#else
#define HAVE_TDM 0
#endif

// ---- helpers --------------------------------------------------------------

__device__ __forceinline__ bf16_t f2bf(float f) { return (bf16_t)f; }  // RNE, hw cvt

__device__ __forceinline__ unsigned pk2bf(float a, float b) {
  v2bf r = {(bf16_t)a, (bf16_t)b};
  return __builtin_bit_cast(unsigned, r);
}

__device__ __forceinline__ v8f wmma_bf16(v16bf a, v16bf b, v8f c) {
  return __builtin_amdgcn_wmma_f32_16x16x32_bf16(false, a, false, b, (short)0, c,
                                                 false, false);
}

// A-fragment (16x32) from a PACKED bf16 row (global mirror or LDS act).
__device__ __forceinline__ v16bf a_chunk_bf(const unsigned* rowd, int lane) {
  const int off = (lane & 16) ? 4 : 0;
  v8u u;
#pragma unroll
  for (int j = 0; j < 4; ++j) {
    u[j]     = rowd[off + j];
    u[4 + j] = rowd[8 + off + j];
  }
  return __builtin_bit_cast(v16bf, u);
}

// A-fragment from an fp32 row (used for agg, produced by f32 atomics).
__device__ __forceinline__ v16bf a_chunk_f32(const float* rowk, int lane) {
  const int kh = (lane & 16) ? 8 : 0;
  const float* p0 = rowk + kh;
  const float* p1 = rowk + 16 + kh;
  v8u u;
#pragma unroll
  for (int j = 0; j < 4; ++j) {
    u[j]     = pk2bf(p0[2 * j], p0[2 * j + 1]);
    u[4 + j] = pk2bf(p1[2 * j], p1[2 * j + 1]);
  }
  return __builtin_bit_cast(v16bf, u);
}

// Stage a layer's packed weights (ndw dwords) into LDS.
// TDM path: one TENSOR_LOAD_TO_LDS issued by wave 0 (1-D tile, 4-byte elems),
// completion on TENSORcnt; the caller's __syncthreads() publishes block-wide.
__device__ __forceinline__ void stage_w(unsigned* dst, const unsigned* __restrict__ src,
                                        int ndw, int tid) {
#if HAVE_TDM
  if (tid < 32) {
    const unsigned lds_addr = (unsigned)(unsigned long long)dst;   // addr[31:0] = LDS offset
    const unsigned long long ga = (unsigned long long)src;
    u32x4 g0;
    g0[0] = 1u;                                        // count=1, user descriptor
    g0[1] = lds_addr;                                  // lds_addr [63:32]
    g0[2] = (unsigned)ga;                              // global_addr [95:64]
    g0[3] = (unsigned)(ga >> 32) | (2u << 30);         // global_addr hi | type=2
    i32x8 g1;
    g1[0] = 2 << 16;                                   // data_size = 4B
    g1[1] = (int)(((unsigned)ndw & 0xFFFFu) << 16);    // tensor_dim0[15:0] @ bits 63:48
    g1[2] = (int)(((unsigned)ndw >> 16) | (1u << 16)); // tensor_dim0 hi | tensor_dim1=1
    g1[3] = (int)((unsigned)ndw << 16);                // tile_dim0 @ bits 127:112
    g1[4] = 1;                                         // tile_dim1 = 1
    g1[5] = ndw;                                       // tensor_dim0_stride low
    g1[6] = 0;
    g1[7] = 0;
    i32x4 g2 = {0, 0, 0, 0}, g3 = {0, 0, 0, 0};
    i32x8 g4 = {0, 0, 0, 0, 0, 0, 0, 0};
    __builtin_amdgcn_tensor_load_to_lds(g0, g1, g2, g3, g4, 0);
    __builtin_amdgcn_s_wait_tensorcnt(0);
  }
#else
  const int n4 = ndw >> 2;
  for (int i = tid; i < n4; i += 256)
    ((uint4*)dst)[i] = ((const uint4*)src)[i];
#endif
}

// One 16xK x Kx128 layer; B-fragments from LDS; bf16 result row into LDS act.
template <int KC, bool RELU>
__device__ __forceinline__ void wmma_layer(const v16bf* afrag, const unsigned* wlds,
                                           const float* __restrict__ bias, int lane,
                                           bf16_t* act_out) {
  const int col = lane & 15, mrow = (lane >> 4) * 8;
#pragma unroll
  for (int nt = 0; nt < 8; ++nt) {
    v8f acc = {};
    const unsigned* base = wlds + (nt * KC) * 256 + lane * 8;
#pragma unroll
    for (int kc = 0; kc < KC; ++kc)
      acc = wmma_bf16(afrag[kc], __builtin_bit_cast(v16bf, *(const v8u*)(base + kc * 256)), acc);
    const float bv = bias[nt * 16 + col];
#pragma unroll
    for (int r = 0; r < 8; ++r) {
      float v = acc[r] + bv;
      if (RELU) v = fmaxf(v, 0.0f);
      act_out[(mrow + r) * ACTW + nt * 16 + col] = f2bf(v);
    }
  }
}

// ---- weight packing: fp32 W[K][128] -> bf16 B-fragment stream -------------
__global__ void pack_b_kernel(const float* __restrict__ W,
                              unsigned* __restrict__ dst, int K) {
  const int idx = blockIdx.x * blockDim.x + threadIdx.x;
  const int total = K * 64;
  if (idx >= total) return;
  const int j    = idx & 7;
  const int lane = (idx >> 3) & 31;
  const int frag = idx >> 8;
  const int KC   = K >> 5;
  const int nt   = frag / KC;
  const int kc   = frag % KC;
  const int n    = nt * 16 + (lane & 15);
  const int k    = kc * 32 + ((j >> 2) << 4) + ((lane >> 4) << 3) + 2 * (j & 3);
  dst[idx] = pk2bf(W[(size_t)k * HD + n], W[(size_t)(k + 1) * HD + n]);
}

// ---- encoder: layer1 (K=3/4) on VALU, layers 2-3 on WMMA ------------------
__global__ __launch_bounds__(256) void encoder_kernel(
    const float* __restrict__ x, int in_dim,
    const float* __restrict__ W1, const float* __restrict__ b1,
    const unsigned* __restrict__ W2p, const float* __restrict__ b2,
    const unsigned* __restrict__ W3p, const float* __restrict__ b3,
    float* __restrict__ outf, unsigned* __restrict__ outb, int ntiles) {
  extern __shared__ __align__(16) char smem[];
  unsigned* wlds = (unsigned*)smem;
  bf16_t* act = (bf16_t*)(smem + 8192 * 4);
  const int tid = threadIdx.x, wave = tid >> 5, lane = tid & 31;
  const int tile = blockIdx.x * 8 + wave;
  const bool active = tile < ntiles;
  bf16_t* buf0 = act + wave * 2 * 16 * ACTW;
  bf16_t* buf1 = buf0 + 16 * ACTW;

  stage_w(wlds, W2p, 8192, tid);
  __builtin_prefetch(W3p + tid * 32, 0, 3);
  if (active) {
    for (int t = lane; t < 16 * HD; t += 32) {
      const int m = t >> 7, n = t & 127;
      const float* xr = x + (size_t)(tile * 16 + m) * in_dim;
      float acc = b1[n];
      for (int k = 0; k < in_dim; ++k) acc = fmaf(xr[k], W1[(size_t)k * HD + n], acc);
      buf0[m * ACTW + n] = f2bf(fmaxf(acc, 0.0f));
    }
  }
  __syncthreads();
  if (active) {
    v16bf a2[4];
#pragma unroll
    for (int kc = 0; kc < 4; ++kc)
      a2[kc] = a_chunk_bf((const unsigned*)buf0 + (lane & 15) * ACTDW + kc * 16, lane);
    wmma_layer<4, true>(a2, wlds, b2, lane, buf1);
  }
  __syncthreads();
  stage_w(wlds, W3p, 8192, tid);
  __syncthreads();
  if (active) {
    v16bf a3[4];
#pragma unroll
    for (int kc = 0; kc < 4; ++kc)
      a3[kc] = a_chunk_bf((const unsigned*)buf1 + (lane & 15) * ACTDW + kc * 16, lane);
    const int col = lane & 15, mrow = (lane >> 4) * 8;
    bf16_t* outbh = (bf16_t*)outb;
#pragma unroll
    for (int nt = 0; nt < 8; ++nt) {
      v8f acc = {};
      const unsigned* base = wlds + (nt * 4) * 256 + lane * 8;
#pragma unroll
      for (int kc = 0; kc < 4; ++kc)
        acc = wmma_bf16(a3[kc], __builtin_bit_cast(v16bf, *(const v8u*)(base + kc * 256)), acc);
      const float bv = b3[nt * 16 + col];
#pragma unroll
      for (int r = 0; r < 8; ++r) {
        const size_t o = (size_t)(tile * 16 + mrow + r) * HD + nt * 16 + col;
        const float v = acc[r] + bv;
        outf[o] = v;
        outbh[o] = f2bf(v);
      }
    }
  }
}

// ---- processor edge MLP: concat(h[src],h[dst],e) 384->128->128->128 + res --
__global__ __launch_bounds__(256) void edge_mlp_kernel(
    const unsigned* __restrict__ hb, float* e, unsigned* eb,
    const int* __restrict__ src, const int* __restrict__ dst,
    const unsigned* __restrict__ W1p, const float* __restrict__ b1,
    const unsigned* __restrict__ W2p, const float* __restrict__ b2,
    const unsigned* __restrict__ W3p, const float* __restrict__ b3, int ntiles) {
  extern __shared__ __align__(16) char smem[];
  unsigned* wlds = (unsigned*)smem;
  bf16_t* act = (bf16_t*)(smem + 24576 * 4);
  const int tid = threadIdx.x, wave = tid >> 5, lane = tid & 31;
  const int tile = blockIdx.x * 8 + wave;
  const bool active = tile < ntiles;
  bf16_t* buf0 = act + wave * 2 * 16 * ACTW;
  bf16_t* buf1 = buf0 + 16 * ACTW;

  stage_w(wlds, W1p, 24576, tid);
  __builtin_prefetch(W2p + tid * 32, 0, 3);

  v16bf a1[12];
  if (active) {
    const int eidx = tile * 16 + (lane & 15);
    const int s = src[eidx], d = dst[eidx];
    const unsigned* hs = hb + (size_t)s * 64;
    const unsigned* hd = hb + (size_t)d * 64;
    const unsigned* er = eb + (size_t)eidx * 64;
#pragma unroll
    for (int kc = 0; kc < 4; ++kc) {
      a1[kc]     = a_chunk_bf(hs + kc * 16, lane);
      a1[4 + kc] = a_chunk_bf(hd + kc * 16, lane);
      a1[8 + kc] = a_chunk_bf(er + kc * 16, lane);
    }
  }
  __syncthreads();
  if (active) wmma_layer<12, true>(a1, wlds, b1, lane, buf0);
  __syncthreads();
  stage_w(wlds, W2p, 8192, tid);
  __builtin_prefetch(W3p + tid * 32, 0, 3);
  __syncthreads();
  if (active) {
    v16bf a2[4];
#pragma unroll
    for (int kc = 0; kc < 4; ++kc)
      a2[kc] = a_chunk_bf((const unsigned*)buf0 + (lane & 15) * ACTDW + kc * 16, lane);
    wmma_layer<4, true>(a2, wlds, b2, lane, buf1);
  }
  __syncthreads();
  stage_w(wlds, W3p, 8192, tid);
  __syncthreads();
  if (active) {
    v16bf a3[4];
#pragma unroll
    for (int kc = 0; kc < 4; ++kc)
      a3[kc] = a_chunk_bf((const unsigned*)buf1 + (lane & 15) * ACTDW + kc * 16, lane);
    const int col = lane & 15, mrow = (lane >> 4) * 8;
    bf16_t* ebh = (bf16_t*)eb;
#pragma unroll
    for (int nt = 0; nt < 8; ++nt) {
      v8f acc = {};
      const unsigned* base = wlds + (nt * 4) * 256 + lane * 8;
#pragma unroll
      for (int kc = 0; kc < 4; ++kc)
        acc = wmma_bf16(a3[kc], __builtin_bit_cast(v16bf, *(const v8u*)(base + kc * 256)), acc);
      const float bv = b3[nt * 16 + col];
#pragma unroll
      for (int r = 0; r < 8; ++r) {
        const size_t o = (size_t)(tile * 16 + mrow + r) * HD + nt * 16 + col;
        const float v = e[o] + acc[r] + bv;   // residual in fp32
        e[o] = v;
        ebh[o] = f2bf(v);
      }
    }
  }
}

// ---- scatter-sum ----------------------------------------------------------
__global__ void zero_kernel(float* __restrict__ p, long long n) {
  const long long i = (long long)blockIdx.x * blockDim.x + threadIdx.x;
  if (i < n) p[i] = 0.0f;
}

__global__ void scatter_kernel(const float* __restrict__ e,
                               const int* __restrict__ dst,
                               float* __restrict__ agg) {
  const long long i = (long long)blockIdx.x * blockDim.x + threadIdx.x;
  if (i >= (long long)EED * HD) return;
  const int edge = (int)(i >> 7);
  const int f    = (int)(i & 127);
  atomicAdd(&agg[(size_t)dst[edge] * HD + f], e[i]);
}

// ---- processor node MLP: concat(h, agg) 256->128->128->128 + residual -----
__global__ __launch_bounds__(256) void node_mlp_kernel(
    float* h, unsigned* hb, const float* __restrict__ agg,
    const unsigned* __restrict__ W1p, const float* __restrict__ b1,
    const unsigned* __restrict__ W2p, const float* __restrict__ b2,
    const unsigned* __restrict__ W3p, const float* __restrict__ b3, int ntiles) {
  extern __shared__ __align__(16) char smem[];
  unsigned* wlds = (unsigned*)smem;
  bf16_t* act = (bf16_t*)(smem + 16384 * 4);
  const int tid = threadIdx.x, wave = tid >> 5, lane = tid & 31;
  const int tile = blockIdx.x * 8 + wave;
  const bool active = tile < ntiles;
  bf16_t* buf0 = act + wave * 2 * 16 * ACTW;
  bf16_t* buf1 = buf0 + 16 * ACTW;

  stage_w(wlds, W1p, 16384, tid);
  __builtin_prefetch(W2p + tid * 32, 0, 3);

  v16bf a1[8];
  if (active) {
    const int row = tile * 16 + (lane & 15);
    const unsigned* hr = hb + (size_t)row * 64;
    const float*    ar = agg + (size_t)row * HD;
#pragma unroll
    for (int kc = 0; kc < 4; ++kc) {
      a1[kc]     = a_chunk_bf(hr + kc * 16, lane);
      a1[4 + kc] = a_chunk_f32(ar + kc * 32, lane);
    }
  }
  __syncthreads();
  if (active) wmma_layer<8, true>(a1, wlds, b1, lane, buf0);
  __syncthreads();
  stage_w(wlds, W2p, 8192, tid);
  __builtin_prefetch(W3p + tid * 32, 0, 3);
  __syncthreads();
  if (active) {
    v16bf a2[4];
#pragma unroll
    for (int kc = 0; kc < 4; ++kc)
      a2[kc] = a_chunk_bf((const unsigned*)buf0 + (lane & 15) * ACTDW + kc * 16, lane);
    wmma_layer<4, true>(a2, wlds, b2, lane, buf1);
  }
  __syncthreads();
  stage_w(wlds, W3p, 8192, tid);
  __syncthreads();
  if (active) {
    v16bf a3[4];
#pragma unroll
    for (int kc = 0; kc < 4; ++kc)
      a3[kc] = a_chunk_bf((const unsigned*)buf1 + (lane & 15) * ACTDW + kc * 16, lane);
    const int col = lane & 15, mrow = (lane >> 4) * 8;
    bf16_t* hbh = (bf16_t*)hb;
#pragma unroll
    for (int nt = 0; nt < 8; ++nt) {
      v8f acc = {};
      const unsigned* base = wlds + (nt * 4) * 256 + lane * 8;
#pragma unroll
      for (int kc = 0; kc < 4; ++kc)
        acc = wmma_bf16(a3[kc], __builtin_bit_cast(v16bf, *(const v8u*)(base + kc * 256)), acc);
      const float bv = b3[nt * 16 + col];
#pragma unroll
      for (int r = 0; r < 8; ++r) {
        const size_t o = (size_t)(tile * 16 + mrow + r) * HD + nt * 16 + col;
        const float v = h[o] + acc[r] + bv;   // residual in fp32
        h[o] = v;
        hbh[o] = f2bf(v);
      }
    }
  }
}

// ---- decoder: 128->128->128 on WMMA, final 128->2 on VALU -----------------
__global__ __launch_bounds__(256) void decoder_kernel(
    const unsigned* __restrict__ hb,
    const unsigned* __restrict__ W1p, const float* __restrict__ b1,
    const unsigned* __restrict__ W2p, const float* __restrict__ b2,
    const float* __restrict__ W3, const float* __restrict__ b3,
    float* __restrict__ out, int ntiles) {
  extern __shared__ __align__(16) char smem[];
  unsigned* wlds = (unsigned*)smem;
  bf16_t* act = (bf16_t*)(smem + 8192 * 4);
  const int tid = threadIdx.x, wave = tid >> 5, lane = tid & 31;
  const int tile = blockIdx.x * 8 + wave;
  const bool active = tile < ntiles;
  bf16_t* buf0 = act + wave * 2 * 16 * ACTW;
  bf16_t* buf1 = buf0 + 16 * ACTW;

  stage_w(wlds, W1p, 8192, tid);
  __builtin_prefetch(W2p + tid * 32, 0, 3);

  v16bf a1[4];
  if (active) {
    const unsigned* hr = hb + (size_t)(tile * 16 + (lane & 15)) * 64;
#pragma unroll
    for (int kc = 0; kc < 4; ++kc) a1[kc] = a_chunk_bf(hr + kc * 16, lane);
  }
  __syncthreads();
  if (active) wmma_layer<4, true>(a1, wlds, b1, lane, buf0);
  __syncthreads();
  stage_w(wlds, W2p, 8192, tid);
  __syncthreads();
  if (active) {
    v16bf a2[4];
#pragma unroll
    for (int kc = 0; kc < 4; ++kc)
      a2[kc] = a_chunk_bf((const unsigned*)buf0 + (lane & 15) * ACTDW + kc * 16, lane);
    wmma_layer<4, true>(a2, wlds, b2, lane, buf1);
  }
  __syncthreads();
  if (active) {
    const int m = lane >> 1, o = lane & 1;
    float acc = b3[o];
    for (int k = 0; k < HD; ++k)
      acc = fmaf((float)buf1[m * ACTW + k], W3[k * OUT_D + o], acc);
    out[(size_t)(tile * 16 + m) * OUT_D + o] = acc;
  }
}

// ---------------------------------------------------------------------------
extern "C" void kernel_launch(void* const* d_in, const int* in_sizes, int n_in,
                              void* d_out, int out_size, void* d_ws, size_t ws_size,
                              hipStream_t stream) {
  const float* x         = (const float*)d_in[0];
  const float* edge_attr = (const float*)d_in[1];
  const int*   eindex    = (const int*)d_in[2];
  const int* src = eindex;
  const int* dst = eindex + EED;

  const float *ne_W1 = (const float*)d_in[3],  *ne_b1 = (const float*)d_in[4];
  const float *ne_W2 = (const float*)d_in[5],  *ne_b2 = (const float*)d_in[6];
  const float *ne_W3 = (const float*)d_in[7],  *ne_b3 = (const float*)d_in[8];
  const float *ee_W1 = (const float*)d_in[9],  *ee_b1 = (const float*)d_in[10];
  const float *ee_W2 = (const float*)d_in[11], *ee_b2 = (const float*)d_in[12];
  const float *ee_W3 = (const float*)d_in[13], *ee_b3 = (const float*)d_in[14];
  const float *pe_W1 = (const float*)d_in[15], *pe_b1 = (const float*)d_in[16];
  const float *pe_W2 = (const float*)d_in[17], *pe_b2 = (const float*)d_in[18];
  const float *pe_W3 = (const float*)d_in[19], *pe_b3 = (const float*)d_in[20];
  const float *pn_W1 = (const float*)d_in[21], *pn_b1 = (const float*)d_in[22];
  const float *pn_W2 = (const float*)d_in[23], *pn_b2 = (const float*)d_in[24];
  const float *pn_W3 = (const float*)d_in[25], *pn_b3 = (const float*)d_in[26];
  const float *nd_W1 = (const float*)d_in[27], *nd_b1 = (const float*)d_in[28];
  const float *nd_W2 = (const float*)d_in[29], *nd_b2 = (const float*)d_in[30];
  const float *nd_W3 = (const float*)d_in[31], *nd_b3 = (const float*)d_in[32];

  // workspace: fp32 masters + bf16 mirrors + packed weights
  char* ws = (char*)d_ws;
  float* h   = (float*)ws;  ws += (size_t)NND * HD * sizeof(float);
  float* e   = (float*)ws;  ws += (size_t)EED * HD * sizeof(float);
  float* agg = (float*)ws;  ws += (size_t)NND * HD * sizeof(float);
  unsigned* hbm = (unsigned*)ws; ws += (size_t)NND * 64 * sizeof(unsigned);
  unsigned* ebm = (unsigned*)ws; ws += (size_t)EED * 64 * sizeof(unsigned);
  unsigned* pw = (unsigned*)ws;

  auto alloc_pw = [&](int K) { unsigned* p = pw; pw += (size_t)K * 64; return p; };
  unsigned* ne_W2p = alloc_pw(128); unsigned* ne_W3p = alloc_pw(128);
  unsigned* ee_W2p = alloc_pw(128); unsigned* ee_W3p = alloc_pw(128);
  unsigned* nd_W1p = alloc_pw(128); unsigned* nd_W2p = alloc_pw(128);
  unsigned *pe_W1p[SSTEP], *pe_W2p[SSTEP], *pe_W3p[SSTEP];
  unsigned *pn_W1p[SSTEP], *pn_W2p[SSTEP], *pn_W3p[SSTEP];
  for (int t = 0; t < SSTEP; ++t) {
    pe_W1p[t] = alloc_pw(384); pe_W2p[t] = alloc_pw(128); pe_W3p[t] = alloc_pw(128);
    pn_W1p[t] = alloc_pw(256); pn_W2p[t] = alloc_pw(128); pn_W3p[t] = alloc_pw(128);
  }

  auto pack = [&](const float* W, unsigned* dstp, int K) {
    int total = K * 64;
    pack_b_kernel<<<(total + 255) / 256, 256, 0, stream>>>(W, dstp, K);
  };
  pack(ne_W2, ne_W2p, 128); pack(ne_W3, ne_W3p, 128);
  pack(ee_W2, ee_W2p, 128); pack(ee_W3, ee_W3p, 128);
  pack(nd_W1, nd_W1p, 128); pack(nd_W2, nd_W2p, 128);
  for (int t = 0; t < SSTEP; ++t) {
    pack(pe_W1 + (size_t)t * 384 * HD, pe_W1p[t], 384);
    pack(pe_W2 + (size_t)t * 128 * HD, pe_W2p[t], 128);
    pack(pe_W3 + (size_t)t * 128 * HD, pe_W3p[t], 128);
    pack(pn_W1 + (size_t)t * 256 * HD, pn_W1p[t], 256);
    pack(pn_W2 + (size_t)t * 128 * HD, pn_W2p[t], 128);
    pack(pn_W3 + (size_t)t * 128 * HD, pn_W3p[t], 128);
  }

  const int ntilesN = NND / 16;   // 3125
  const int ntilesE = EED / 16;   // 25000
  const int nblkN = (ntilesN + 7) / 8;
  const int nblkE = (ntilesE + 7) / 8;
  const size_t actB = (size_t)8 * 2 * 16 * ACTW * sizeof(bf16_t);  // 69632
  const size_t shEnc  = 8192 * 4 + actB;
  const size_t shEdge = 24576 * 4 + actB;
  const size_t shNode = 16384 * 4 + actB;
  const size_t shDec  = 8192 * 4 + actB;

  encoder_kernel<<<nblkN, 256, shEnc, stream>>>(x, NODE_IN, ne_W1, ne_b1,
      ne_W2p, ne_b2, ne_W3p, ne_b3, h, hbm, ntilesN);
  encoder_kernel<<<nblkE, 256, shEnc, stream>>>(edge_attr, EDGE_IN, ee_W1, ee_b1,
      ee_W2p, ee_b2, ee_W3p, ee_b3, e, ebm, ntilesE);

  const long long aggN = (long long)NND * HD;
  const long long scatN = (long long)EED * HD;
  for (int t = 0; t < SSTEP; ++t) {
    edge_mlp_kernel<<<nblkE, 256, shEdge, stream>>>(
        hbm, e, ebm, src, dst,
        pe_W1p[t], pe_b1 + (size_t)t * HD,
        pe_W2p[t], pe_b2 + (size_t)t * HD,
        pe_W3p[t], pe_b3 + (size_t)t * HD, ntilesE);
    zero_kernel<<<(unsigned)((aggN + 255) / 256), 256, 0, stream>>>(agg, aggN);
    scatter_kernel<<<(unsigned)((scatN + 255) / 256), 256, 0, stream>>>(e, dst, agg);
    node_mlp_kernel<<<nblkN, 256, shNode, stream>>>(
        h, hbm, agg,
        pn_W1p[t], pn_b1 + (size_t)t * HD,
        pn_W2p[t], pn_b2 + (size_t)t * HD,
        pn_W3p[t], pn_b3 + (size_t)t * HD, ntilesN);
  }

  decoder_kernel<<<nblkN, 256, shDec, stream>>>(hbm, nd_W1p, nd_b1, nd_W2p, nd_b2,
                                                nd_W3, nd_b3, (float*)d_out, ntilesN);
}
